// Adapter_37529424232596
// MI455X (gfx1250) — matrix operand order repaired and checked by reading, not compiled, not run on hardware
//
#include <hip/hip_runtime.h>
#include <hip/hip_bf16.h>
#include <stdint.h>

// ---------------------------------------------------------------------------
// Problem constants (fixed by the reference): B=32, T=8, H=W=14, C=768, CA=384
// ---------------------------------------------------------------------------
#define BT        256            // B*T
#define LTOK      197            // H*W + 1
#define NTOK      196            // H*W
#define CDIM      768
#define CADIM     384
#define MROWS     (BT * NTOK)    // 50176

typedef __attribute__((ext_vector_type(16))) __bf16 v16bf;
typedef __attribute__((ext_vector_type(8)))  float  v8f;
typedef __attribute__((ext_vector_type(4)))  int    v4i;

__device__ __forceinline__ unsigned short f2bf(float f) {
    unsigned int u = __float_as_uint(f);
    u += 0x7fffu + ((u >> 16) & 1u);      // round-to-nearest-even
    return (unsigned short)(u >> 16);
}

__device__ __forceinline__ v8f wmma_bf16(v16bf a, v16bf b, v8f c) {
    return __builtin_amdgcn_wmma_f32_16x16x32_bf16(
        /*neg_a=*/false, a, /*neg_b=*/false, b,
        /*c_mod=*/(short)0, c, /*reuse_a=*/false, /*reuse_b=*/false);
}

// Async global -> LDS copy of 16 bytes (per lane).  ASYNCcnt-tracked DMA path:
// no VGPR staging, overlaps with WMMA on the current LDS buffer.
// Builtin signature: (int4 addrspace(1)* gaddr, int4 addrspace(3)* ldsaddr,
//                     imm offset, imm cpol)
__device__ __forceinline__ void async_cp16(const unsigned short* g, unsigned short* l) {
#if __has_builtin(__builtin_amdgcn_global_load_async_to_lds_b128)
    __builtin_amdgcn_global_load_async_to_lds_b128(
        (__attribute__((address_space(1))) v4i*)(uintptr_t)g,
        (__attribute__((address_space(3))) v4i*)(unsigned int)(uintptr_t)l,
        0, 0);
#else
    unsigned int laddr = (unsigned int)(uintptr_t)l;      // LDS offset = low 32 bits
    unsigned long long gaddr = (unsigned long long)(uintptr_t)g;
    asm volatile("global_load_async_to_lds_b128 %0, %1, off"
                 :: "v"(laddr), "v"(gaddr) : "memory");
#endif
}

__device__ __forceinline__ void wait_async0() {
#if __has_builtin(__builtin_amdgcn_s_wait_asynccnt)
    __builtin_amdgcn_s_wait_asynccnt(0);
#else
    asm volatile("s_wait_asynccnt 0x0" ::: "memory");
#endif
}

// A fragment (16-bit A 16x32): lanes 0-15 row M=lane, K {0..7,16..23};
// lanes 16-31 row M=lane-16, K {8..15,24..31}. LDS tile is [row][32] bf16.
__device__ __forceinline__ v16bf load_fragA(const unsigned short* lds, int row, int lhalf) {
    union { uint4 q[2]; v16bf v; } u;
    const int base = row * 32 + lhalf * 8;
    u.q[0] = *(const uint4*)(lds + base);
    u.q[1] = *(const uint4*)(lds + base + 16);
    return u.v;
}

// B fragment (16-bit B 32x16): lanes 0-15 col N=lane, K {0..7,16..23};
// lanes 16-31 col N=lane-16, K {8..15,24..31}. LDS tile is [col][32] bf16 (N-major).
__device__ __forceinline__ v16bf load_fragB(const unsigned short* lds, int row, int lhalf) {
    union { uint4 q[2]; v16bf v; } u;
    const int base = row * 32 + lhalf * 8;
    u.q[0] = *(const uint4*)(lds + base);
    u.q[1] = *(const uint4*)(lds + base + 16);
    return u.v;
}

// ---------------------------------------------------------------------------
// Prep: W1 [384x768] and W2 [768x384] -> bf16 (both are already N-major, so
// they serve directly as WMMA B operands: out[m,n] = sum_k A[m,k]*W[n,k]).
// ---------------------------------------------------------------------------
__global__ void convert_weights_k(const float* __restrict__ W1,
                                  const float* __restrict__ W2,
                                  unsigned short* __restrict__ W1b,
                                  unsigned short* __restrict__ W2b) {
    int i = blockIdx.x * 256 + threadIdx.x;   // 0 .. 294911
    W1b[i] = f2bf(W1[i]);
    W2b[i] = f2bf(W2[i]);
}

// out[:,0,:] = x[:,0,:]   (CLS passthrough)
__global__ void cls_copy_k(const float* __restrict__ x, float* __restrict__ out) {
    int i = blockIdx.x * 256 + threadIdx.x;   // 0 .. 196607
    int bt = i / CDIM, c = i % CDIM;
    size_t off = (size_t)bt * LTOK * CDIM + c;
    out[off] = x[off];
}

// ---------------------------------------------------------------------------
// fc1: h1[m, n] = sum_k x_row(m)[k] * W1[n, k] + b1[n]
// Block: 256 threads (8 waves). Tile 128(M) x 64(N), K-step 32, double-buffered
// LDS. B tile via async global->LDS DMA; A tile f32->bf16 VALU convert (both
// for tile k+1, overlapped with WMMAs on tile k).
// ---------------------------------------------------------------------------
__global__ void fc1_wmma_k(const float* __restrict__ x,
                           const unsigned short* __restrict__ W1b,
                           const float* __restrict__ b1,
                           float* __restrict__ h1) {
    __shared__ unsigned short lA[2][128 * 32];   // 2 x 8 KB
    __shared__ unsigned short lB[2][64 * 32];    // 2 x 4 KB

    const int tid  = threadIdx.x;
    const int lane = tid & 31;
    const int wid  = tid >> 5;
    const int wm   = (wid >> 1) * 32;
    const int wn   = (wid & 1) * 32;
    const int lrow  = lane & 15;
    const int lhalf = lane >> 4;

    const int m0 = blockIdx.x * 128;
    const int n0 = blockIdx.y * 64;

    // cooperative-load coordinates
    const int arow = tid >> 1;                // 0..127
    const int aseg = (tid & 1) * 16;          // bf16 elems
    const int am   = m0 + arow;
    const float* ap = x + ((size_t)(am / NTOK) * LTOK + 1 + (am % NTOK)) * CDIM;

    const int brow = tid >> 2;                // 0..63
    const int bseg = (tid & 3) * 8;
    const unsigned short* bp = W1b + (size_t)(n0 + brow) * CDIM;

    v8f acc[2][2] = {};

    // tile loaders ---------------------------------------------------------
    auto load_tiles = [&](int k0, int buf) {
        // B: async DMA, 16 bytes/thread
        async_cp16(bp + k0 + bseg, &lB[buf][brow * 32 + bseg]);
        // A: 16 f32 -> 16 bf16 per thread
        const float4* a4 = (const float4*)(ap + k0 + aseg);
        union { unsigned short s[16]; uint4 q[2]; } pk;
#pragma unroll
        for (int j = 0; j < 4; ++j) {
            float4 f = a4[j];
            pk.s[j * 4 + 0] = f2bf(f.x);
            pk.s[j * 4 + 1] = f2bf(f.y);
            pk.s[j * 4 + 2] = f2bf(f.z);
            pk.s[j * 4 + 3] = f2bf(f.w);
        }
        *(uint4*)&lA[buf][arow * 32 + aseg]     = pk.q[0];
        *(uint4*)&lA[buf][arow * 32 + aseg + 8] = pk.q[1];
    };

    load_tiles(0, 0);
    wait_async0();
    __syncthreads();

    const int NK = CDIM / 32;
    for (int kt = 0; kt < NK; ++kt) {
        const int cur = kt & 1;
        if (kt + 1 < NK) load_tiles((kt + 1) * 32, cur ^ 1);

        const unsigned short* cA = lA[cur];
        const unsigned short* cB = lB[cur];
        v16bf a0  = load_fragA(cA, wm + lrow,      lhalf);
        v16bf a1  = load_fragA(cA, wm + 16 + lrow, lhalf);
        v16bf bb0 = load_fragB(cB, wn + lrow,      lhalf);
        v16bf bb1 = load_fragB(cB, wn + 16 + lrow, lhalf);

        acc[0][0] = wmma_bf16(a0, bb0, acc[0][0]);
        acc[0][1] = wmma_bf16(a0, bb1, acc[0][1]);
        acc[1][0] = wmma_bf16(a1, bb0, acc[1][0]);
        acc[1][1] = wmma_bf16(a1, bb1, acc[1][1]);

        wait_async0();
        __syncthreads();
    }

    // store: D layout -> lane 0-15: N=lane, M=vgpr; lane 16-31: M=vgpr+8
#pragma unroll
    for (int tm = 0; tm < 2; ++tm) {
#pragma unroll
        for (int tn = 0; tn < 2; ++tn) {
            const int ng = n0 + wn + tn * 16 + lrow;
            const float bn = b1[ng];
            const int mbase = m0 + wm + tm * 16 + lhalf * 8;
            float* op = h1 + (size_t)mbase * CADIM + ng;
#pragma unroll
            for (int v = 0; v < 8; ++v)
                op[(size_t)v * CADIM] = acc[tm][tn][v] + bn;
        }
    }
}

// ---------------------------------------------------------------------------
// Depthwise 3x3x3 conv over (t, y, x) with zero padding; emits bf16 for fc2.
// Block: 128 threads = 128 consecutive channels of one spatial row.
// ---------------------------------------------------------------------------
__global__ void conv_dw_k(const float* __restrict__ h1,
                          const float* __restrict__ cw,
                          const float* __restrict__ cb,
                          unsigned short* __restrict__ h2) {
    const int r  = blockIdx.x / 3;                     // 0 .. 50175
    const int ca = (blockIdx.x % 3) * 128 + threadIdx.x;

    const int bt = r / NTOK, p = r % NTOK;
    const int y  = p / 14,   xw = p % 14;
    const int b  = bt >> 3,  tt = bt & 7;

    float sum = cb[ca];
    const float* w = cw + ca * 27;
#pragma unroll
    for (int dt = -1; dt <= 1; ++dt) {
        const int t2 = tt + dt;
        if ((unsigned)t2 >= 8u) continue;
#pragma unroll
        for (int dy = -1; dy <= 1; ++dy) {
            const int y2 = y + dy;
            if ((unsigned)y2 >= 14u) continue;
#pragma unroll
            for (int dx = -1; dx <= 1; ++dx) {
                const int x2 = xw + dx;
                if ((unsigned)x2 >= 14u) continue;
                const int r2 = ((b * 8 + t2) * NTOK + y2 * 14 + x2);
                sum += w[(dt + 1) * 9 + (dy + 1) * 3 + (dx + 1)] *
                       h1[(size_t)r2 * CADIM + ca];
            }
        }
    }
    h2[(size_t)r * CADIM + ca] = f2bf(sum);
}

// ---------------------------------------------------------------------------
// fc2: out_row(m)[n] = sum_k h2[m,k]*W2[n,k] + b2[n] + x_row(m)[n]
// Same tiling as fc1; both A and B tiles via async global->LDS DMA,
// double-buffered; residual + CLS-skip scatter fused into the epilogue.
// ---------------------------------------------------------------------------
__global__ void fc2_wmma_k(const unsigned short* __restrict__ h2,
                           const unsigned short* __restrict__ W2b,
                           const float* __restrict__ b2,
                           const float* __restrict__ x,
                           float* __restrict__ out) {
    __shared__ unsigned short lA[2][128 * 32];
    __shared__ unsigned short lB[2][64 * 32];

    const int tid  = threadIdx.x;
    const int lane = tid & 31;
    const int wid  = tid >> 5;
    const int wm   = (wid >> 1) * 32;
    const int wn   = (wid & 1) * 32;
    const int lrow  = lane & 15;
    const int lhalf = lane >> 4;

    const int m0 = blockIdx.x * 128;
    const int n0 = blockIdx.y * 64;

    const int arow = tid >> 1;
    const int aseg = (tid & 1) * 16;
    const unsigned short* ap = h2 + (size_t)(m0 + arow) * CADIM;

    const int brow = tid >> 2;
    const int bseg = (tid & 3) * 8;
    const unsigned short* bp = W2b + (size_t)(n0 + brow) * CADIM;

    v8f acc[2][2] = {};

    auto load_tiles = [&](int k0, int buf) {
        async_cp16(ap + k0 + aseg,     &lA[buf][arow * 32 + aseg]);
        async_cp16(ap + k0 + aseg + 8, &lA[buf][arow * 32 + aseg + 8]);
        async_cp16(bp + k0 + bseg,     &lB[buf][brow * 32 + bseg]);
    };

    load_tiles(0, 0);
    wait_async0();
    __syncthreads();

    const int NK = CADIM / 32;
    for (int kt = 0; kt < NK; ++kt) {
        const int cur = kt & 1;
        if (kt + 1 < NK) load_tiles((kt + 1) * 32, cur ^ 1);

        const unsigned short* cA = lA[cur];
        const unsigned short* cB = lB[cur];
        v16bf a0  = load_fragA(cA, wm + lrow,      lhalf);
        v16bf a1  = load_fragA(cA, wm + 16 + lrow, lhalf);
        v16bf bb0 = load_fragB(cB, wn + lrow,      lhalf);
        v16bf bb1 = load_fragB(cB, wn + 16 + lrow, lhalf);

        acc[0][0] = wmma_bf16(a0, bb0, acc[0][0]);
        acc[0][1] = wmma_bf16(a0, bb1, acc[0][1]);
        acc[1][0] = wmma_bf16(a1, bb0, acc[1][0]);
        acc[1][1] = wmma_bf16(a1, bb1, acc[1][1]);

        wait_async0();
        __syncthreads();
    }

#pragma unroll
    for (int tm = 0; tm < 2; ++tm) {
#pragma unroll
        for (int tn = 0; tn < 2; ++tn) {
            const int ng = n0 + wn + tn * 16 + lrow;
            const float bn = b2[ng];
            const int mbase = m0 + wm + tm * 16 + lhalf * 8;
#pragma unroll
            for (int v = 0; v < 8; ++v) {
                const int mg = mbase + v;
                const int bt = mg / NTOK, l = mg % NTOK;
                const size_t off = ((size_t)bt * LTOK + 1 + l) * CDIM + ng;
                out[off] = acc[tm][tn][v] + bn + x[off];
            }
        }
    }
}

// ---------------------------------------------------------------------------
// Launch
// ---------------------------------------------------------------------------
extern "C" void kernel_launch(void* const* d_in, const int* in_sizes, int n_in,
                              void* d_out, int out_size, void* d_ws, size_t ws_size,
                              hipStream_t stream) {
    (void)in_sizes; (void)n_in; (void)out_size; (void)ws_size;

    const float* x  = (const float*)d_in[0];
    const float* W1 = (const float*)d_in[1];
    const float* b1 = (const float*)d_in[2];
    const float* cw = (const float*)d_in[3];
    const float* cb = (const float*)d_in[4];
    const float* W2 = (const float*)d_in[5];
    const float* b2 = (const float*)d_in[6];
    float* out = (float*)d_out;

    // workspace layout (all 16B-aligned offsets)
    char* ws = (char*)d_ws;
    float*          h1  = (float*)(ws);                                  // 77,070,336 B
    unsigned short* h2  = (unsigned short*)(ws + 77070336);              // 38,535,168 B
    unsigned short* W1b = (unsigned short*)(ws + 77070336 + 38535168);   //    589,824 B
    unsigned short* W2b = (unsigned short*)(ws + 77070336 + 38535168 + 589824);

    convert_weights_k<<<dim3(1152), dim3(256), 0, stream>>>(W1, W2, W1b, W2b);
    cls_copy_k<<<dim3(768), dim3(256), 0, stream>>>(x, out);
    fc1_wmma_k<<<dim3(MROWS / 128, CADIM / 64), dim3(256), 0, stream>>>(x, W1b, b1, h1);
    conv_dw_k<<<dim3(MROWS * 3), dim3(128), 0, stream>>>(h1, cw, cb, h2);
    fc2_wmma_k<<<dim3(MROWS / 128, CDIM / 64), dim3(256), 0, stream>>>(h2, W2b, b2, x, out);
}
